// MultiHeadedAttentionSANM_893353197808
// MI455X (gfx1250) — compile-verified, matrix operand order checked by reading, not run on hardware
//
#include <hip/hip_runtime.h>
#include <hip/hip_bf16.h>

typedef __attribute__((ext_vector_type(8)))  float  v8f;
typedef __attribute__((ext_vector_type(16))) __bf16 v16bf;
typedef __attribute__((ext_vector_type(4)))  int    v4i;

#define B_   8
#define T_   2048
#define D_   512
#define H_   4
#define DK_  128
#define KC_  11
#define N3_  1536
#define BT_  (B_ * T_)

#if __has_builtin(__builtin_amdgcn_global_load_async_to_lds_b128) && \
    __has_builtin(__builtin_amdgcn_s_wait_asynccnt)
#define USE_ASYNC_LDS 1
#else
#define USE_ASYNC_LDS 0
#endif

typedef __attribute__((address_space(1))) v4i* gm_v4i_p;   // global (AS1)
typedef __attribute__((address_space(3))) v4i* lds_v4i_p;  // LDS (AS3)

// 16B global -> LDS copy. Async (ASYNCcnt-tracked) when available.
__device__ __forceinline__ void cp_b128(unsigned short* lds_dst, const unsigned short* gsrc) {
#if USE_ASYNC_LDS
  __builtin_amdgcn_global_load_async_to_lds_b128((gm_v4i_p)gsrc, (lds_v4i_p)lds_dst, 0, 0);
#else
  *reinterpret_cast<uint4*>(lds_dst) = *reinterpret_cast<const uint4*>(gsrc);
#endif
}
__device__ __forceinline__ void cp_wait() {
#if USE_ASYNC_LDS
  __builtin_amdgcn_s_wait_asynccnt(0);
#endif
}

__device__ __forceinline__ unsigned short f2bf(float f) {
  unsigned u = __float_as_uint(f);
  unsigned r = u + 0x7FFFu + ((u >> 16) & 1u);   // round-to-nearest-even
  return (unsigned short)(r >> 16);
}
__device__ __forceinline__ float bf2f(unsigned short h) {
  return __uint_as_float(((unsigned)h) << 16);
}

union FragU { unsigned u[8]; v16bf v; };

// Gather a 16x32 bf16 A/B fragment per CDNA5 ISA 7.12.2:
// lane holds row/col (lane&15); dword j holds K pair:
//   j<4 : K = half*8 + 2j ; j>=4 : K = 16 + half*8 + 2(j-4)
// The 8 dwords form two contiguous 16B runs -> lowers to 2x ds_load_b128.
__device__ __forceinline__ v16bf gather_frag(const unsigned short* base, int row, int ld) {
  const int lane = threadIdx.x & 31;
  const int half = lane >> 4;
  const int l16  = lane & 15;
  const unsigned short* p = base + (row + l16) * ld;
  FragU f;
#pragma unroll
  for (int j = 0; j < 8; ++j) {
    const int kb = (j < 4) ? (half * 8 + 2 * j) : (16 + half * 8 + 2 * (j - 4));
    f.u[j] = *(const unsigned*)(p + kb);
  }
  return f.v;
}

// ---------------------------------------------------------------- converts
__global__ void f32_to_bf16_kernel(const float* __restrict__ in,
                                   unsigned short* __restrict__ out, int n) {
  int i = blockIdx.x * blockDim.x + threadIdx.x;
  if (i < n) out[i] = f2bf(in[i]);
}

// in: K x N row-major (f32) -> out: N x K row-major (bf16)
__global__ void transpose_f32_to_bf16_kernel(const float* __restrict__ in,
                                             unsigned short* __restrict__ out,
                                             int K, int N) {
  int i = blockIdx.x * blockDim.x + threadIdx.x;
  if (i < K * N) {
    int n = i / K, k = i % K;
    out[n * K + k] = f2bf(in[k * N + n]);
  }
}

// ---------------------------------------------------------------- GEMM core
#define AS_SZ (64 * 32)
#define BS_SZ (128 * 32)

__device__ __forceinline__ void stage_ab(const unsigned short* __restrict__ A,
                                         const unsigned short* __restrict__ BT,
                                         int K, int rowA0, int rowB0, int k0,
                                         unsigned short* As, unsigned short* Bs) {
  const int tid = threadIdx.x;
  {   // A tile 64x32 (row-major, stride 32)
    int row = tid >> 2, jc = (tid & 3) * 8;
    cp_b128(As + row * 32 + jc, A + (rowA0 + row) * K + k0 + jc);
  }
  {   // B tile 128x32 N-major (from pre-transposed BT)
    int row = tid >> 1, jc = (tid & 1) * 16;
    cp_b128(Bs + row * 32 + jc,     BT + (rowB0 + row) * K + k0 + jc);
    cp_b128(Bs + row * 32 + jc + 8, BT + (rowB0 + row) * K + k0 + jc + 8);
  }
}

// C(64x128 block) = A(rowA0..+64, :) @ BT(rowB0..+128, :)^T ; double-buffered LDS,
// one barrier per K-step; stage k+1 overlaps WMMAs on k.
__device__ __forceinline__ void gemm_tile_core(
    const unsigned short* __restrict__ A, const unsigned short* __restrict__ BT,
    int K, int rowA0, int rowB0,
    unsigned short* As, unsigned short* Bs, v8f acc[4]) {
  const int wid = threadIdx.x >> 5;
  const int wm = wid & 3, wn = wid >> 2;
  stage_ab(A, BT, K, rowA0, rowB0, 0, As, Bs);
  int p = 0;
  for (int k0 = 0; k0 < K; k0 += 32, p ^= 1) {
    cp_wait();
    __syncthreads();
    if (k0 + 32 < K)
      stage_ab(A, BT, K, rowA0, rowB0, k0 + 32, As + (p ^ 1) * AS_SZ, Bs + (p ^ 1) * BS_SZ);
    const unsigned short* Asb = As + p * AS_SZ;
    const unsigned short* Bsb = Bs + p * BS_SZ;
    v16bf a = gather_frag(Asb, wm * 16, 32);
#pragma unroll
    for (int n = 0; n < 4; ++n) {
      v16bf b = gather_frag(Bsb, wn * 64 + n * 16, 32);
      acc[n] = __builtin_amdgcn_wmma_f32_16x16x32_bf16(false, a, false, b,
                                                       (short)0, acc[n], false, false);
    }
  }
}

// ---------------------------------------------------------------- QKV GEMM
__global__ __launch_bounds__(256) void qkv_gemm_kernel(
    const unsigned short* __restrict__ x_bf, const unsigned short* __restrict__ wqkvT,
    const float* __restrict__ bqkv, unsigned short* __restrict__ qkv_bf) {
  __shared__ __attribute__((aligned(16))) unsigned short As[2 * AS_SZ];
  __shared__ __attribute__((aligned(16))) unsigned short Bs[2 * BS_SZ];
  v8f acc[4] = {};
  const int rowA0 = blockIdx.x * 64;
  const int colN0 = blockIdx.y * 128;
  gemm_tile_core(x_bf, wqkvT, 512, rowA0, colN0, As, Bs, acc);

  const int lane = threadIdx.x & 31, wid = threadIdx.x >> 5;
  const int half = lane >> 4, l16 = lane & 15;
  const int wm = wid & 3, wn = wid >> 2;
#pragma unroll
  for (int n = 0; n < 4; ++n) {
    const int col = colN0 + wn * 64 + n * 16 + l16;
    const float scale = (col < D_) ? 0.08838834764831845f : 1.0f;  // DK^-0.5 on q
    const float bias = bqkv[col];
#pragma unroll
    for (int r = 0; r < 8; ++r) {
      const int row = rowA0 + wm * 16 + r + 8 * half;
      qkv_bf[row * N3_ + col] = f2bf((acc[n][r] + bias) * scale);
    }
  }
}

// ---------------------------------------------------------------- FSMN conv
__global__ void fsmn_kernel(const unsigned short* __restrict__ qkv_bf,
                            const float* __restrict__ mask,
                            const float* __restrict__ w,
                            float* __restrict__ fsmn) {
  int i = blockIdx.x * blockDim.x + threadIdx.x;   // over B*T*D
  if (i >= BT_ * D_) return;
  const int d = i % D_;
  const int bt = i / D_;
  const int t = bt % T_;
  const int b = bt / T_;
  float acc = 0.f;
#pragma unroll
  for (int k = 0; k < KC_; ++k) {
    const int tin = t + k - 5;                     // LPAD = 5
    if (tin >= 0 && tin < T_) {
      const float m = mask[b * T_ + tin];
      const float v = bf2f(qkv_bf[(b * T_ + tin) * N3_ + 2 * D_ + d]);
      acc += v * m * w[d * KC_ + k];
    }
  }
  const float mt = mask[b * T_ + t];
  const float vt = bf2f(qkv_bf[(b * T_ + t) * N3_ + 2 * D_ + d]);
  fsmn[i] = (acc + vt * mt) * mt;
}

// ---------------------------------------------------------------- attention
#define KS_SZ (64 * 136)   // [key][d], stride 136 (d-pairs contiguous)
#define VT_SZ (128 * 66)   // [d][key], stride 66 (key-pairs contiguous)

__device__ __forceinline__ void stage_kv(const unsigned short* __restrict__ qkv_bf,
                                         const float* __restrict__ mask,
                                         int b, int h, int kb,
                                         unsigned short* Ksb, unsigned short* Vtb,
                                         float* maskSb) {
  const int tid = threadIdx.x;
  {   // K block: async 16B copies into Ks[key][d]
    int row = tid >> 2, cb = (tid & 3) * 32;
    const unsigned short* g = qkv_bf + (b * T_ + kb + row) * N3_ + D_ + h * DK_ + cb;
#pragma unroll
    for (int i = 0; i < 4; ++i) cp_b128(Ksb + row * 136 + cb + i * 8, g + i * 8);
  }
  {   // V block transposed on store: Vt[d][key]
    int t = tid >> 2, db = (tid & 3) * 32;
    const unsigned* g = reinterpret_cast<const unsigned*>(
        qkv_bf + (b * T_ + kb + t) * N3_ + 2 * D_ + h * DK_ + db);
#pragma unroll
    for (int j = 0; j < 16; ++j) {
      unsigned u = g[j];
      int d = db + 2 * j;
      Vtb[d * 66 + t]       = (unsigned short)(u & 0xFFFFu);
      Vtb[(d + 1) * 66 + t] = (unsigned short)(u >> 16);
    }
  }
  if (tid < 64) maskSb[tid] = mask[b * T_ + kb + tid];
}

// grid: (T/128, B*H). 8 waves/block; wave owns a 16-row q strip, DK=128.
__global__ __launch_bounds__(256) void attention_kernel(
    const unsigned short* __restrict__ qkv_bf,
    const float* __restrict__ mask,
    unsigned short* __restrict__ ctx_bf) {
  __shared__ __attribute__((aligned(16))) unsigned short Ks[2 * KS_SZ];
  __shared__ __attribute__((aligned(16))) unsigned short Vt[2 * VT_SZ];
  __shared__ __attribute__((aligned(16))) unsigned short Pl[8 * 16 * 66];
  __shared__ float maskS[2 * 64];

  const int tid = threadIdx.x, lane = tid & 31, wid = tid >> 5;
  const int half = lane >> 4, l16 = lane & 15;
  const int b = blockIdx.y / H_, h = blockIdx.y % H_;
  const int q0 = blockIdx.x * 128 + wid * 16;

  // Q fragments (q pre-scaled by DK^-0.5): 4 K-steps of 32, resident in VGPRs
  const unsigned short* qbase = qkv_bf + (b * T_ + q0) * N3_ + h * DK_;
  v16bf qf[4];
#pragma unroll
  for (int kk = 0; kk < 4; ++kk) qf[kk] = gather_frag(qbase + kk * 32, 0, N3_);

  v8f o[8] = {};
  float m_run[8], l_run[8];
#pragma unroll
  for (int r = 0; r < 8; ++r) { m_run[r] = -1e30f; l_run[r] = 0.f; }

  unsigned short* Pw = Pl + wid * (16 * 66);

  stage_kv(qkv_bf, mask, b, h, 0, Ks, Vt, maskS);
  int p = 0;
  for (int kb = 0; kb < T_; kb += 64, p ^= 1) {
    cp_wait();
    __syncthreads();                       // buffer p ready; buffer p^1 free
    if (kb + 64 < T_)
      stage_kv(qkv_bf, mask, b, h, kb + 64,
               Ks + (p ^ 1) * KS_SZ, Vt + (p ^ 1) * VT_SZ, maskS + (p ^ 1) * 64);
    const unsigned short* Ksb = Ks + p * KS_SZ;
    const unsigned short* Vtb = Vt + p * VT_SZ;
    const float* maskSb = maskS + p * 64;

    // S = Q @ K^T : 16x64 strip = 4 tiles x 4 K-steps
    v8f s4[4] = {};
#pragma unroll
    for (int kk = 0; kk < 4; ++kk) {
#pragma unroll
      for (int t = 0; t < 4; ++t) {
        v16bf bfr = gather_frag(Ksb + kk * 32, t * 16, 136);
        s4[t] = __builtin_amdgcn_wmma_f32_16x16x32_bf16(false, qf[kk], false, bfr,
                                                        (short)0, s4[t], false, false);
      }
    }
    // key-column mask
    float mcol[4];
#pragma unroll
    for (int t = 0; t < 4; ++t) {
      mcol[t] = maskSb[t * 16 + l16];
      if (mcol[t] == 0.f) {
#pragma unroll
        for (int r = 0; r < 8; ++r) s4[t][r] = -1e30f;
      }
    }
    // online softmax: C-layout replicates row M=r+8*half across a 16-lane half,
    // so row reductions use shfl_xor masks 1,2,4,8 only.
    float mnew[8], alpha[8];
#pragma unroll
    for (int r = 0; r < 8; ++r) {
      float rm = s4[0][r];
#pragma unroll
      for (int t = 1; t < 4; ++t) rm = fmaxf(rm, s4[t][r]);
#pragma unroll
      for (int off = 1; off < 16; off <<= 1) rm = fmaxf(rm, __shfl_xor(rm, off, 32));
      mnew[r] = fmaxf(m_run[r], rm);
      alpha[r] = __expf(m_run[r] - mnew[r]);
      m_run[r] = mnew[r];
    }
#pragma unroll
    for (int r = 0; r < 8; ++r) {
      float rs = 0.f;
#pragma unroll
      for (int t = 0; t < 4; ++t) {
        float pv = (mcol[t] == 0.f) ? 0.f : __expf(s4[t][r] - mnew[r]);
        rs += pv;
        Pw[(r + 8 * half) * 66 + t * 16 + l16] = f2bf(pv);
      }
#pragma unroll
      for (int off = 1; off < 16; off <<= 1) rs += __shfl_xor(rs, off, 32);
      l_run[r] = l_run[r] * alpha[r] + rs;
#pragma unroll
      for (int n = 0; n < 8; ++n) o[n][r] = o[n][r] * alpha[r];
    }
    // O += P @ V : K=64 -> 2 WMMA K-steps, 8 d-tiles
#pragma unroll
    for (int kk = 0; kk < 2; ++kk) {
      v16bf a = gather_frag(Pw + kk * 32, 0, 66);
#pragma unroll
      for (int n = 0; n < 8; ++n) {
        v16bf bfr = gather_frag(Vtb + kk * 32, n * 16, 66);
        o[n] = __builtin_amdgcn_wmma_f32_16x16x32_bf16(false, a, false, bfr,
                                                       (short)0, o[n], false, false);
      }
    }
  }

  // normalize and store ctx (bf16)
#pragma unroll
  for (int r = 0; r < 8; ++r) {
    const float inv = (l_run[r] > 0.f) ? (1.0f / l_run[r]) : 0.f;
    const int row = b * T_ + q0 + r + 8 * half;
#pragma unroll
    for (int n = 0; n < 8; ++n)
      ctx_bf[row * D_ + h * DK_ + n * 16 + l16] = f2bf(o[n][r] * inv);
  }
}

// ---------------------------------------------------------------- out proj
__global__ __launch_bounds__(256) void out_gemm_kernel(
    const unsigned short* __restrict__ ctx_bf, const unsigned short* __restrict__ woutT,
    const float* __restrict__ bout, const float* __restrict__ fsmn,
    float* __restrict__ out) {
  __shared__ __attribute__((aligned(16))) unsigned short As[2 * AS_SZ];
  __shared__ __attribute__((aligned(16))) unsigned short Bs[2 * BS_SZ];
  v8f acc[4] = {};
  const int rowA0 = blockIdx.x * 64;
  const int colN0 = blockIdx.y * 128;
  gemm_tile_core(ctx_bf, woutT, 512, rowA0, colN0, As, Bs, acc);

  const int lane = threadIdx.x & 31, wid = threadIdx.x >> 5;
  const int half = lane >> 4, l16 = lane & 15;
  const int wm = wid & 3, wn = wid >> 2;
#pragma unroll
  for (int n = 0; n < 4; ++n) {
    const int col = colN0 + wn * 64 + n * 16 + l16;
    const float bias = bout[col];
#pragma unroll
    for (int r = 0; r < 8; ++r) {
      const int row = rowA0 + wm * 16 + r + 8 * half;
      out[row * D_ + col] = acc[n][r] + bias + fsmn[row * D_ + col];
    }
  }
}

// ---------------------------------------------------------------- launch
extern "C" void kernel_launch(void* const* d_in, const int* in_sizes, int n_in,
                              void* d_out, int out_size, void* d_ws, size_t ws_size,
                              hipStream_t stream) {
  (void)in_sizes; (void)n_in; (void)out_size; (void)ws_size;
  const float* x      = (const float*)d_in[0];
  const float* mask   = (const float*)d_in[1];
  const float* Wqkv   = (const float*)d_in[2];
  const float* bqkv   = (const float*)d_in[3];
  const float* Wout   = (const float*)d_in[4];
  const float* bout   = (const float*)d_in[5];
  const float* fsmn_w = (const float*)d_in[6];
  float* out = (float*)d_out;

  char* ws = (char*)d_ws;
  unsigned short* x_bf   = (unsigned short*)(ws);                 // 16 MB
  unsigned short* wqkvT  = (unsigned short*)(ws + 16777216);      // 1.5 MB
  unsigned short* woutT  = (unsigned short*)(ws + 18350080);      // 0.5 MB
  unsigned short* qkv_bf = (unsigned short*)(ws + 18874368);      // 48 MB
  unsigned short* ctx_bf = (unsigned short*)(ws + 69206016);      // 16 MB
  float*          fsmn   = (float*)(ws + 85983232);               // 32 MB

  // 1) precision conversion / weight transposes
  f32_to_bf16_kernel<<<(BT_ * D_) / 256, 256, 0, stream>>>(x, x_bf, BT_ * D_);
  transpose_f32_to_bf16_kernel<<<(D_ * N3_) / 256, 256, 0, stream>>>(Wqkv, wqkvT, D_, N3_);
  transpose_f32_to_bf16_kernel<<<(D_ * D_) / 256, 256, 0, stream>>>(Wout, woutT, D_, D_);

  // 2) qkv = x @ Wqkv + bqkv (q pre-scaled)
  qkv_gemm_kernel<<<dim3(BT_ / 64, N3_ / 128), 256, 0, stream>>>(x_bf, wqkvT, bqkv, qkv_bf);

  // 3) FSMN depthwise conv on v
  fsmn_kernel<<<(BT_ * D_) / 256, 256, 0, stream>>>(qkv_bf, mask, fsmn_w, fsmn);

  // 4) flash attention -> ctx
  attention_kernel<<<dim3(T_ / 128, B_ * H_), 256, 0, stream>>>(qkv_bf, mask, ctx_bf);

  // 5) out = ctx @ Wout + bout + fsmn
  out_gemm_kernel<<<dim3(BT_ / 64, D_ / 128), 256, 0, stream>>>(ctx_bf, woutT, bout, fsmn, out);
}
